// SequentialModel_730144440611
// MI455X (gfx1250) — matrix operand order repaired and checked by reading, not compiled, not run on hardware
//
#include <hip/hip_runtime.h>

#define NB   64    // batch
#define NT   1024  // time steps
#define DIN  256   // input dim
#define NH   512   // hidden dim
#define NOUT 128   // output dim

typedef __bf16 bf16;
typedef __bf16 bf16x16 __attribute__((ext_vector_type(16)));
typedef float  f32x8   __attribute__((ext_vector_type(8)));
typedef unsigned int u32x4 __attribute__((ext_vector_type(4)));
typedef unsigned int u32x8 __attribute__((ext_vector_type(8)));

__device__ __forceinline__ bf16x16 make_frag(u32x4 q0, u32x4 q1) {
  u32x8 w;
  w[0] = q0[0]; w[1] = q0[1]; w[2] = q0[2]; w[3] = q0[3];
  w[4] = q1[0]; w[5] = q1[1]; w[6] = q1[2]; w[7] = q1[3];
  return __builtin_bit_cast(bf16x16, w);
}

// A fragment: 16x32 bf16 (M rows striped on lanes).
// lane l: m = l&15, half = l>>4. VGPR0-3 = K[half*8 .. half*8+7],
// VGPR4-7 = K[16+half*8 .. 16+half*8+7]  (ISA 7.12.2 16-bit A layout)
__device__ __forceinline__ bf16x16 frag_a(const bf16* base, size_t row_stride, int kb) {
  const int lane = threadIdx.x & 31;
  const int m = lane & 15, hf = lane >> 4;
  const bf16* p = base + (size_t)m * row_stride + (size_t)(kb + hf * 8);
  u32x4 q0 = *(const u32x4*)p;
  u32x4 q1 = *(const u32x4*)(p + 16);   // +16 elems = +32 bytes
  return make_frag(q0, q1);
}

// B fragment: 32x16, B[k][n] = W[n][k] with W row-major (rows = output cols).
// lane l: n = l&15, half = l>>4, holds K[half*8 .. half*8+7], K[16+half*8 .. 16+half*8+7].
__device__ __forceinline__ bf16x16 frag_b(const bf16* rowbase, int row_stride, int kb) {
  const int lane = threadIdx.x & 31;
  const int n = lane & 15, hf = lane >> 4;
  const bf16* p = rowbase + (size_t)n * row_stride + (size_t)(kb + hf * 8);
  u32x4 q0 = *(const u32x4*)p;
  u32x4 q1 = *(const u32x4*)(p + 16);   // +16 elems = +32 bytes
  return make_frag(q0, q1);
}

__device__ __forceinline__ f32x8 wmma_bf16(bf16x16 a, bf16x16 b, f32x8 c) {
  return __builtin_amdgcn_wmma_f32_16x16x32_bf16(false, a, false, b, (short)0, c,
                                                 false, false);
}

__device__ __forceinline__ float sigf(float x) { return 1.0f / (1.0f + __expf(-x)); }

__device__ __forceinline__ void grid_barrier(unsigned* slot, unsigned nwg) {
  __syncthreads();
  if (threadIdx.x == 0) {
    __threadfence();                      // release h_t stores to device scope
    atomicAdd(slot, 1u);
    while (__hip_atomic_load(slot, __ATOMIC_RELAXED, __HIP_MEMORY_SCOPE_AGENT) < nwg) {
      __builtin_amdgcn_s_sleep(1);
    }
  }
  __syncthreads();
  __threadfence();                        // acquire: invalidate near caches
}

// ---------------------------------------------------------------------------
__global__ void init_kernel(const float* __restrict__ bih0, const float* __restrict__ bhh0,
                            const float* __restrict__ bih1, const float* __restrict__ bhh1,
                            float* __restrict__ bsum0, float* __restrict__ bsum1,
                            bf16* __restrict__ zeropg,
                            unsigned* __restrict__ bar0, unsigned* __restrict__ bar1) {
  const int i0 = blockIdx.x * blockDim.x + threadIdx.x;
  const int stride = gridDim.x * blockDim.x;
  for (int j = i0; j < 4 * NH; j += stride) {
    bsum0[j] = bih0[j] + bhh0[j];
    bsum1[j] = bih1[j] + bhh1[j];
  }
  for (int j = i0; j < NB * NH; j += stride) zeropg[j] = (bf16)0.0f;
  for (int j = i0; j < NT; j += stride) { bar0[j] = 0u; bar1[j] = 0u; }
}

__global__ void cvt_kernel(const float* __restrict__ in, bf16* __restrict__ out, int n) {
  int i = blockIdx.x * blockDim.x + threadIdx.x;
  const int stride = gridDim.x * blockDim.x;
  for (; i < n; i += stride) out[i] = (bf16)in[i];
}

// ---------------------------------------------------------------------------
// Layer 0: g = x_t @ Wih0^T + h_{t-1} @ Whh0^T + bsum.  32 WGs x 128 thr.
// WG owns 16 h-cols (64 gate cols); wave = one 16-batch M-tile, 4 gate tiles.
__global__ __launch_bounds__(128, 1) void lstm0_kernel(
    const bf16* __restrict__ xbf,    // [NB][NT][DIN] bf16
    const float* __restrict__ Wih,   // [4H][DIN] f32
    const float* __restrict__ Whh,   // [4H][NH]  f32
    const float* __restrict__ bsum,  // [4H] f32
    const bf16* __restrict__ zeropg, // [NB*NH] zeros
    bf16* __restrict__ hs0,          // [NT][NB][NH] bf16 out
    unsigned* __restrict__ bar) {
  __shared__ __align__(16) bf16 sWih[64 * (DIN + 8)];
  __shared__ __align__(16) bf16 sWhh[64 * (NH + 8)];
  const int j0 = blockIdx.x * 16;

  for (int idx = threadIdx.x; idx < 64 * DIN; idx += blockDim.x) {
    const int c = idx >> 8, k = idx & (DIN - 1);
    const int grow = (c >> 4) * NH + j0 + (c & 15);
    sWih[c * (DIN + 8) + k] = (bf16)Wih[(size_t)grow * DIN + k];
  }
  for (int idx = threadIdx.x; idx < 64 * NH; idx += blockDim.x) {
    const int c = idx >> 9, k = idx & (NH - 1);
    const int grow = (c >> 4) * NH + j0 + (c & 15);
    sWhh[c * (NH + 8) + k] = (bf16)Whh[(size_t)grow * NH + k];
  }
  __syncthreads();

  const int lane = threadIdx.x & 31;
  const int b0 = (threadIdx.x >> 5) * 16;   // M-tile base (batch)
  const int n = lane & 15, half = lane >> 4;

  float bias[4];
#pragma unroll
  for (int g = 0; g < 4; ++g) bias[g] = bsum[g * NH + j0 + n];

  f32x8 cstate;
#pragma unroll
  for (int r = 0; r < 8; ++r) cstate[r] = 0.0f;

  for (int t = 0; t < NT; ++t) {
    f32x8 acc[4];
#pragma unroll
    for (int g = 0; g < 4; ++g)
#pragma unroll
      for (int r = 0; r < 8; ++r) acc[g][r] = 0.0f;

    // input contribution: A = x[b][t][*], row stride NT*DIN between batches
    const bf16* xbase = xbf + ((size_t)b0 * NT + (size_t)t) * DIN;
#pragma unroll 4
    for (int kb = 0; kb < DIN; kb += 32) {
      bf16x16 a = frag_a(xbase, (size_t)NT * DIN, kb);
#pragma unroll
      for (int g = 0; g < 4; ++g) {
        bf16x16 b = frag_b(&sWih[(g * 16) * (DIN + 8)], DIN + 8, kb);
        acc[g] = wmma_bf16(a, b, acc[g]);
      }
    }
    // recurrent contribution: A = h_{t-1}
    const bf16* hprev = (t == 0) ? zeropg : (hs0 + (size_t)(t - 1) * NB * NH);
    const bf16* hbase = hprev + (size_t)b0 * NH;
#pragma unroll 4
    for (int kb = 0; kb < NH; kb += 32) {
      bf16x16 a = frag_a(hbase, NH, kb);
#pragma unroll
      for (int g = 0; g < 4; ++g) {
        bf16x16 b = frag_b(&sWhh[(g * 16) * (NH + 8)], NH + 8, kb);
        acc[g] = wmma_bf16(a, b, acc[g]);
      }
    }
    // gates + state update (C/D layout: VGPR r -> M = r + 8*half, N = lane&15)
    bf16* hout = hs0 + (size_t)t * NB * NH;
#pragma unroll
    for (int r = 0; r < 8; ++r) {
      const float gi = sigf(acc[0][r] + bias[0]);
      const float gf = sigf(acc[1][r] + bias[1]);
      const float gg = tanhf(acc[2][r] + bias[2]);
      const float go = sigf(acc[3][r] + bias[3]);
      float cn = gf * cstate[r] + gi * gg;
      cn = fminf(fmaxf(cn, -1.0f), 1.0f);
      cstate[r] = cn;
      const float hn = go * tanhf(cn);
      const int m = b0 + r + half * 8;
      hout[(size_t)m * NH + j0 + n] = (bf16)hn;
    }
    grid_barrier(bar + t, gridDim.x);
  }
}

// ---------------------------------------------------------------------------
// Layer 1: g = hs0_t @ Wih1^T + h_{t-1} @ Whh1^T + bsum.  Ping-pong h buffers.
__global__ __launch_bounds__(128, 1) void lstm1_kernel(
    const bf16* __restrict__ hs0,    // [NT][NB][NH]
    const float* __restrict__ Wih,   // [4H][NH]
    const float* __restrict__ Whh,   // [4H][NH]
    const float* __restrict__ bsum,
    const bf16* __restrict__ zeropg,
    bf16* __restrict__ hbuf,         // [2][NB*NH]
    unsigned* __restrict__ bar) {
  __shared__ __align__(16) bf16 sWih[64 * (NH + 8)];
  __shared__ __align__(16) bf16 sWhh[64 * (NH + 8)];
  const int j0 = blockIdx.x * 16;

  for (int idx = threadIdx.x; idx < 64 * NH; idx += blockDim.x) {
    const int c = idx >> 9, k = idx & (NH - 1);
    const int grow = (c >> 4) * NH + j0 + (c & 15);
    sWih[c * (NH + 8) + k] = (bf16)Wih[(size_t)grow * NH + k];
    sWhh[c * (NH + 8) + k] = (bf16)Whh[(size_t)grow * NH + k];
  }
  __syncthreads();

  const int lane = threadIdx.x & 31;
  const int b0 = (threadIdx.x >> 5) * 16;
  const int n = lane & 15, half = lane >> 4;

  float bias[4];
#pragma unroll
  for (int g = 0; g < 4; ++g) bias[g] = bsum[g * NH + j0 + n];

  f32x8 cstate;
#pragma unroll
  for (int r = 0; r < 8; ++r) cstate[r] = 0.0f;

  for (int t = 0; t < NT; ++t) {
    f32x8 acc[4];
#pragma unroll
    for (int g = 0; g < 4; ++g)
#pragma unroll
      for (int r = 0; r < 8; ++r) acc[g][r] = 0.0f;

    const bf16* a1 = hs0 + ((size_t)t * NB + b0) * NH;     // layer-0 output
#pragma unroll 4
    for (int kb = 0; kb < NH; kb += 32) {
      bf16x16 a = frag_a(a1, NH, kb);
#pragma unroll
      for (int g = 0; g < 4; ++g) {
        bf16x16 b = frag_b(&sWih[(g * 16) * (NH + 8)], NH + 8, kb);
        acc[g] = wmma_bf16(a, b, acc[g]);
      }
    }
    const bf16* hprev = (t == 0) ? zeropg : (hbuf + (size_t)((t - 1) & 1) * NB * NH);
    const bf16* hbase = hprev + (size_t)b0 * NH;
#pragma unroll 4
    for (int kb = 0; kb < NH; kb += 32) {
      bf16x16 a = frag_a(hbase, NH, kb);
#pragma unroll
      for (int g = 0; g < 4; ++g) {
        bf16x16 b = frag_b(&sWhh[(g * 16) * (NH + 8)], NH + 8, kb);
        acc[g] = wmma_bf16(a, b, acc[g]);
      }
    }
    bf16* hout = hbuf + (size_t)(t & 1) * NB * NH;
#pragma unroll
    for (int r = 0; r < 8; ++r) {
      const float gi = sigf(acc[0][r] + bias[0]);
      const float gf = sigf(acc[1][r] + bias[1]);
      const float gg = tanhf(acc[2][r] + bias[2]);
      const float go = sigf(acc[3][r] + bias[3]);
      float cn = gf * cstate[r] + gi * gg;
      cn = fminf(fmaxf(cn, -1.0f), 1.0f);
      cstate[r] = cn;
      const float hn = go * tanhf(cn);
      const int m = b0 + r + half * 8;
      hout[(size_t)m * NH + j0 + n] = (bf16)hn;
    }
    grid_barrier(bar + t, gridDim.x);
  }
}

// ---------------------------------------------------------------------------
// Head: out = h_last @ Wc^T + bc.  One WG, wave = M-tile, 8 N-tiles each.
__global__ __launch_bounds__(128, 1) void head_kernel(
    const bf16* __restrict__ hlast,  // [NB][NH]
    const bf16* __restrict__ wc,     // [NOUT][NH] bf16
    const float* __restrict__ bc,    // [NOUT]
    float* __restrict__ out) {       // [NB][NOUT]
  const int lane = threadIdx.x & 31;
  const int b0 = (threadIdx.x >> 5) * 16;
  const int n = lane & 15, half = lane >> 4;

  f32x8 acc[8];
#pragma unroll
  for (int nt = 0; nt < 8; ++nt)
#pragma unroll
    for (int r = 0; r < 8; ++r) acc[nt][r] = 0.0f;

#pragma unroll 4
  for (int kb = 0; kb < NH; kb += 32) {
    bf16x16 a = frag_a(hlast + (size_t)b0 * NH, NH, kb);
#pragma unroll
    for (int nt = 0; nt < 8; ++nt) {
      bf16x16 b = frag_b(wc + (size_t)(nt * 16) * NH, NH, kb);
      acc[nt] = wmma_bf16(a, b, acc[nt]);
    }
  }
#pragma unroll
  for (int nt = 0; nt < 8; ++nt) {
    const float bias = bc[nt * 16 + n];
#pragma unroll
    for (int r = 0; r < 8; ++r) {
      const int m = b0 + r + half * 8;
      out[(size_t)m * NOUT + nt * 16 + n] = acc[nt][r] + bias;
    }
  }
}

// ---------------------------------------------------------------------------
extern "C" void kernel_launch(void* const* d_in, const int* in_sizes, int n_in,
                              void* d_out, int out_size, void* d_ws, size_t ws_size,
                              hipStream_t stream) {
  const float* x    = (const float*)d_in[0];
  const float* Wih0 = (const float*)d_in[1];
  const float* Whh0 = (const float*)d_in[2];
  const float* bih0 = (const float*)d_in[3];
  const float* bhh0 = (const float*)d_in[4];
  const float* Wih1 = (const float*)d_in[5];
  const float* Whh1 = (const float*)d_in[6];
  const float* bih1 = (const float*)d_in[7];
  const float* bhh1 = (const float*)d_in[8];
  const float* Wc   = (const float*)d_in[9];
  const float* bc   = (const float*)d_in[10];

  char* ws = (char*)d_ws;
  size_t off = 0;
  auto take = [&](size_t bytes) -> char* {
    char* p = ws + off;
    off = (off + bytes + 255) & ~(size_t)255;
    return p;
  };
  bf16*     xbf   = (bf16*)take((size_t)NB * NT * DIN * 2);   // 32 MiB
  bf16*     hs0   = (bf16*)take((size_t)NT * NB * NH * 2);    // 64 MiB
  bf16*     hbuf  = (bf16*)take((size_t)2 * NB * NH * 2);
  bf16*     zpg   = (bf16*)take((size_t)NB * NH * 2);
  bf16*     wcbf  = (bf16*)take((size_t)NOUT * NH * 2);
  float*    bsum0 = (float*)take((size_t)4 * NH * 4);
  float*    bsum1 = (float*)take((size_t)4 * NH * 4);
  unsigned* bar0  = (unsigned*)take((size_t)NT * 4);
  unsigned* bar1  = (unsigned*)take((size_t)NT * 4);
  (void)ws_size; (void)in_sizes; (void)n_in; (void)out_size;

  init_kernel<<<32, 256, 0, stream>>>(bih0, bhh0, bih1, bhh1, bsum0, bsum1, zpg,
                                      bar0, bar1);
  cvt_kernel<<<2048, 256, 0, stream>>>(x, xbf, NB * NT * DIN);
  cvt_kernel<<<64, 256, 0, stream>>>(Wc, wcbf, NOUT * NH);
  lstm0_kernel<<<32, 128, 0, stream>>>(xbf, Wih0, Whh0, bsum0, zpg, hs0, bar0);
  lstm1_kernel<<<32, 128, 0, stream>>>(hs0, Wih1, Whh1, bsum1, zpg, hbuf, bar1);
  head_kernel<<<1, 128, 0, stream>>>(hbuf + (size_t)NB * NH, wcbf, bc,
                                     (float*)d_out);
}
